// GATSPEmbedder_21062519620286
// MI455X (gfx1250) — matrix operand order repaired and checked by reading, not compiled, not run on hardware
//
#include <hip/hip_runtime.h>
#include <hip/hip_bf16.h>
#include <cstdint>

// Problem constants (from reference)
#define F_DIM   64
#define H_HEADS 2
#define D_DIM   128      // H*F
#define NPG_C   50
#define NEG_ATTN 0.2f
#define NEG_ACT  0.1f

typedef __attribute__((ext_vector_type(16))) __bf16 bf16x16;
typedef __attribute__((ext_vector_type(8)))  float  floatx8;

union BFrag { bf16x16 v; unsigned int u[8]; unsigned short s[16]; };

__device__ __forceinline__ unsigned short f2bf(float x) {
    unsigned int u = __float_as_uint(x);
    unsigned int r = (u + 0x7FFFu + ((u >> 16) & 1u)) >> 16;  // RNE
    return (unsigned short)r;
}

__device__ __forceinline__ void atomicMaxF(float* addr, float val) {
    // monotonic-encoding trick; addr must be initialized to -inf
    if (val >= 0.0f) atomicMax((int*)addr, __float_as_int(val));
    else             atomicMin((unsigned int*)addr, __float_as_uint(val));
}

// ---------------- conversion kernels ----------------

__global__ void k_f32_to_bf16(const float* __restrict__ x, unsigned short* __restrict__ y, int n) {
    int t = blockIdx.x * blockDim.x + threadIdx.x;
    if (t < n) y[t] = f2bf(x[t]);
}

// W[K][128] (row major f32) -> Wt[128][K] bf16 (so B fragments are k-contiguous)
__global__ void k_weight_tr_bf16(const float* __restrict__ W, unsigned short* __restrict__ Wt, int K) {
    int t = blockIdx.x * blockDim.x + threadIdx.x;
    if (t >= K * D_DIM) return;
    int k = t / D_DIM, n = t % D_DIM;
    Wt[n * K + k] = f2bf(W[t]);
}

__global__ void k_zero(float* __restrict__ p, int n) {
    int t = blockIdx.x * blockDim.x + threadIdx.x;
    if (t < n) p[t] = 0.0f;
}

__global__ void k_init_layer(float* __restrict__ m, float* __restrict__ denom,
                             float* __restrict__ rst, int nh, int nd) {
    int t = blockIdx.x * blockDim.x + threadIdx.x;
    if (t < nh) { m[t] = -INFINITY; denom[t] = 0.0f; }
    if (t < nd) rst[t] = 0.0f;
}

// ---------------- WMMA GEMM: C[N,128] = A[N,K](bf16) x Wt(bf16, [128][K]) ----------------
// 8 waves/block, one 16-row tile per wave; weights staged in LDS via async copy.
__global__ __launch_bounds__(256) void k_gemm_wmma(const unsigned short* __restrict__ A,
                                                   const unsigned short* __restrict__ Bt,
                                                   float* __restrict__ C, int K, int Nrows) {
    __shared__ unsigned short sW[D_DIM * D_DIM];   // 32 KB max (Wt[128][K] bf16)

    const int tid  = threadIdx.x;
    const int lane = tid & 31;
    const int hlf  = lane >> 4;        // lane half (0/1)
    const int l15  = lane & 15;

    // ---- stage Wt into LDS with CDNA5 async global->LDS copies (ASYNCcnt) ----
    {
        typedef __attribute__((address_space(3))) unsigned short lds_us;
        lds_us* lw = (lds_us*)sW;
        const int nchunks = (D_DIM * K * 2) / 16;  // 16B chunks: 1024 (K=64) / 2048 (K=128)
        for (int idx = tid; idx < nchunks; idx += 256) {   // uniform trip count per wave
            unsigned loff = (unsigned)(unsigned long long)(lw + idx * 8);
            const unsigned short* g = Bt + idx * 8;
            asm volatile("global_load_async_to_lds_b128 %0, %1, off"
                         :: "v"(loff), "v"(g) : "memory");
        }
        asm volatile("s_wait_asynccnt 0x0" ::: "memory");  // my chunks landed in LDS
        __syncthreads();                                   // publish to all waves
    }

    const int wave = blockIdx.x * (blockDim.x >> 5) + (tid >> 5);
    const int row0 = wave * 16;
    if (row0 >= Nrows) return;         // uniform per wave -> EXEC all ones for WMMA

    floatx8 acc[8];
    #pragma unroll
    for (int nt = 0; nt < 8; ++nt)
        #pragma unroll
        for (int i = 0; i < 8; ++i) acc[nt][i] = 0.0f;

    const int m = row0 + l15;          // A row held by this lane
    for (int kt = 0; kt < K; kt += 32) {
        // A fragment: 16-bit A 16x32 layout. lane<16: k = {0..7,16..23}; lane>=16: +8.
        BFrag af;
        const unsigned short* arow = A + (size_t)m * K + kt;
        #pragma unroll
        for (int j = 0; j < 8; ++j) {
            int kk = ((j < 4) ? 0 : 16) + hlf * 8 + ((j & 3) * 2);  // first k of the pair
            af.u[j] = *(const unsigned int*)(arow + kk);
        }
        if (kt + 32 < K) __builtin_prefetch(arow + 32, 0, 3);       // global_prefetch_b8

        #pragma unroll
        for (int nt = 0; nt < 8; ++nt) {
            // B fragment from LDS: k = {0..7,16..23} (+8 for lane>=16) -> 2x ds_load_b128
            BFrag bf;
            const uint4* bp = (const uint4*)(sW + (size_t)(nt * 16 + l15) * K + kt + hlf * 8);
            uint4 lo = bp[0], hi = bp[2];
            bf.u[0] = lo.x; bf.u[1] = lo.y; bf.u[2] = lo.z; bf.u[3] = lo.w;
            bf.u[4] = hi.x; bf.u[5] = hi.y; bf.u[6] = hi.z; bf.u[7] = hi.w;
            acc[nt] = __builtin_amdgcn_wmma_f32_16x16x32_bf16(
                false, af.v, false, bf.v, (short)0, acc[nt], false, false);
        }
    }

    // D layout: VGPR v -> row = row0 + v + 8*hlf, col = nt*16 + l15
    #pragma unroll
    for (int nt = 0; nt < 8; ++nt) {
        int col = nt * 16 + l15;
        #pragma unroll
        for (int v = 0; v < 8; ++v) {
            int row = row0 + v + 8 * hlf;
            C[(size_t)row * D_DIM + col] = acc[nt][v];
        }
    }
}

// ---------------- attention coefficients: el/er [N,H] ----------------
__global__ void k_attn_coef(const float* __restrict__ feat, const float* __restrict__ al,
                            const float* __restrict__ ar, float* __restrict__ el,
                            float* __restrict__ er, int nh) {
    int t = blockIdx.x * blockDim.x + threadIdx.x;
    if (t >= nh) return;
    int h = t & 1, n = t >> 1;
    const float* fr  = feat + (size_t)n * D_DIM + h * F_DIM;
    const float* alh = al + h * F_DIM;
    const float* arh = ar + h * F_DIM;
    float sl = 0.0f, sr = 0.0f;
    #pragma unroll 8
    for (int f = 0; f < F_DIM; ++f) { float x = fr[f]; sl += x * alh[f]; sr += x * arh[f]; }
    el[t] = sl; er[t] = sr;
}

// ---------------- edge softmax passes ----------------
__global__ void k_edge_logits(const int* __restrict__ src, const int* __restrict__ dst,
                              const float* __restrict__ el, const float* __restrict__ er,
                              float* __restrict__ eArr, float* __restrict__ mmax, int EH) {
    int t = blockIdx.x * blockDim.x + threadIdx.x;
    if (t >= EH) return;
    int h = t & 1, e = t >> 1;
    int s = src[e], d = dst[e];
    float v = el[s * H_HEADS + h] + er[d * H_HEADS + h];
    v = (v > 0.0f) ? v : NEG_ATTN * v;
    eArr[t] = v;
    atomicMaxF(&mmax[d * H_HEADS + h], v);
}

__global__ void k_edge_exp(const int* __restrict__ dst, const float* __restrict__ mmax,
                           float* __restrict__ eArr, float* __restrict__ denom, int EH) {
    int t = blockIdx.x * blockDim.x + threadIdx.x;
    if (t >= EH) return;
    int h = t & 1, e = t >> 1;
    int d = dst[e];
    float ex = __expf(eArr[t] - mmax[d * H_HEADS + h]);
    eArr[t] = ex;
    atomicAdd(&denom[d * H_HEADS + h], ex);
}

// one wave per edge; lane covers 4 columns (float4), accumulate unnormalized sum
__global__ __launch_bounds__(256) void k_edge_aggregate(const int* __restrict__ src,
                                                        const int* __restrict__ dst,
                                                        const float* __restrict__ eArr,
                                                        const float* __restrict__ feat,
                                                        float* __restrict__ rst, int E) {
    int t = blockIdx.x * blockDim.x + threadIdx.x;
    int e = t >> 5, lane = t & 31;
    if (e >= E) return;
    int s = src[e], d = dst[e];
    int c = lane * 4, h = c >> 6;
    float w = eArr[e * H_HEADS + h];
    const float4 f4 = *(const float4*)(feat + (size_t)s * D_DIM + c);
    float* o = rst + (size_t)d * D_DIM + c;
    atomicAdd(o + 0, w * f4.x); atomicAdd(o + 1, w * f4.y);
    atomicAdd(o + 2, w * f4.z); atomicAdd(o + 3, w * f4.w);
}

// ---------------- norm + residual + act + pooling (one wave per node) ----------------
__global__ __launch_bounds__(256) void k_post_norm(const float* __restrict__ rst,
                                                   const float* __restrict__ denom,
                                                   const float* res,  // may alias hout
                                                   const float* __restrict__ g,
                                                   const float* __restrict__ b,
                                                   float* hout, unsigned short* __restrict__ hbf,
                                                   float* __restrict__ pool, int layer, int Nn) {
    int wave = blockIdx.x * (blockDim.x >> 5) + (threadIdx.x >> 5);
    int lane = threadIdx.x & 31;
    if (wave >= Nn) return;
    int n = wave, c = lane * 4, h = c >> 6;
    float inv = 1.0f / denom[n * H_HEADS + h];
    float4 r4 = *(const float4*)(rst + (size_t)n * D_DIM + c);
    float4 s4 = *(const float4*)(res + (size_t)n * D_DIM + c);
    float x[4] = { r4.x * inv + s4.x, r4.y * inv + s4.y, r4.z * inv + s4.z, r4.w * inv + s4.w };

    float s1 = x[0] + x[1] + x[2] + x[3];
    float s2 = x[0]*x[0] + x[1]*x[1] + x[2]*x[2] + x[3]*x[3];
    #pragma unroll
    for (int off = 16; off > 0; off >>= 1) {
        s1 += __shfl_xor(s1, off, 32);
        s2 += __shfl_xor(s2, off, 32);
    }
    float mu  = s1 * (1.0f / 128.0f);
    float var = s2 * (1.0f / 128.0f) - mu * mu;
    float rstd = rsqrtf(var + 1e-5f);

    float act[4];
    #pragma unroll
    for (int i = 0; i < 4; ++i) {
        float y = (x[i] - mu) * rstd * g[c + i] + b[c + i];
        act[i] = (y > 0.0f) ? y : NEG_ACT * y;
        hout[(size_t)n * D_DIM + c + i] = act[i];
        hbf [(size_t)n * D_DIM + c + i] = f2bf(act[i]);
    }
    // head-mean + graph-mean pooling
    float part[4];
    #pragma unroll
    for (int i = 0; i < 4; ++i) part[i] = __shfl_xor(act[i], 16, 32);
    if (lane < 16) {
        int gidx = n / NPG_C;
        float* po = pool + (size_t)gidx * 192 + layer * F_DIM + c;
        #pragma unroll
        for (int i = 0; i < 4; ++i) atomicAdd(po + i, 0.5f * (act[i] + part[i]));
    }
}

__global__ void k_final(const float* __restrict__ pool, float* __restrict__ out, int n) {
    int t = blockIdx.x * blockDim.x + threadIdx.x;
    if (t >= n) return;
    float v = pool[t] * (1.0f / (float)NPG_C);
    out[t] = (v > 0.0f) ? v : NEG_ACT * v;
}

// ---------------- host side ----------------
extern "C" void kernel_launch(void* const* d_in, const int* in_sizes, int n_in,
                              void* d_out, int out_size, void* d_ws, size_t ws_size,
                              hipStream_t stream) {
    const int N = in_sizes[0] / F_DIM;        // 102400
    const int E = in_sizes[17];               // 921600
    const int Bg = N / NPG_C;                 // 2048

    const float* node_feats = (const float*)d_in[0];
    const float* Wl[3]  = { (const float*)d_in[1],  (const float*)d_in[7],  (const float*)d_in[12] };
    const float* all_[3]= { (const float*)d_in[2],  (const float*)d_in[8],  (const float*)d_in[13] };
    const float* arl[3] = { (const float*)d_in[3],  (const float*)d_in[9],  (const float*)d_in[14] };
    const float* resW0  =   (const float*)d_in[4];
    const float* gl[3]  = { (const float*)d_in[5],  (const float*)d_in[10], (const float*)d_in[15] };
    const float* bl[3]  = { (const float*)d_in[6],  (const float*)d_in[11], (const float*)d_in[16] };
    const int* src = (const int*)d_in[17];
    const int* dst = (const int*)d_in[18];
    float* out = (float*)d_out;

    // workspace carve-up (256B aligned)
    char* p = (char*)d_ws;
    auto carve = [&](size_t bytes) { void* r = (void*)p; p += (bytes + 255) & ~(size_t)255; return r; };
    unsigned short* hbf = (unsigned short*)carve((size_t)N * D_DIM * 2);
    unsigned short* Wt  = (unsigned short*)carve((size_t)D_DIM * D_DIM * 2);
    float* hres = (float*)carve((size_t)N * D_DIM * 4);   // residual in, activation out
    float* feat = (float*)carve((size_t)N * D_DIM * 4);
    float* rst  = (float*)carve((size_t)N * D_DIM * 4);
    float* el   = (float*)carve((size_t)N * H_HEADS * 4);
    float* er   = (float*)carve((size_t)N * H_HEADS * 4);
    float* mmax = (float*)carve((size_t)N * H_HEADS * 4);
    float* den  = (float*)carve((size_t)N * H_HEADS * 4);
    float* eArr = (float*)carve((size_t)E * H_HEADS * 4);
    float* pool = (float*)carve((size_t)Bg * 192 * 4);

    auto cdiv = [](int a, int b) { return (a + b - 1) / b; };
    const int TB = 256;

    k_zero<<<cdiv(Bg * 192, TB), TB, 0, stream>>>(pool, Bg * 192);
    // initial bf16 activations = node_feats [N,64]
    k_f32_to_bf16<<<cdiv(N * F_DIM, TB), TB, 0, stream>>>(node_feats, hbf, N * F_DIM);

    for (int l = 0; l < 3; ++l) {
        const int K = (l == 0) ? F_DIM : D_DIM;
        const int gemmBlocks = cdiv(N, 16 * 8);   // 8 waves/block, 16 rows/wave

        k_weight_tr_bf16<<<cdiv(K * D_DIM, TB), TB, 0, stream>>>(Wl[l], Wt, K);
        k_gemm_wmma<<<gemmBlocks, TB, 0, stream>>>(hbf, Wt, feat, K, N);

        if (l == 0) { // residual = node_feats @ resW0, written into hres
            k_weight_tr_bf16<<<cdiv(K * D_DIM, TB), TB, 0, stream>>>(resW0, Wt, K);
            k_gemm_wmma<<<gemmBlocks, TB, 0, stream>>>(hbf, Wt, hres, K, N);
        } // for l>0, hres already holds previous layer activation (identity residual)

        k_attn_coef<<<cdiv(N * H_HEADS, TB), TB, 0, stream>>>(feat, all_[l], arl[l], el, er, N * H_HEADS);
        k_init_layer<<<cdiv(N * D_DIM, TB), TB, 0, stream>>>(mmax, den, rst, N * H_HEADS, N * D_DIM);
        k_edge_logits<<<cdiv(E * H_HEADS, TB), TB, 0, stream>>>(src, dst, el, er, eArr, mmax, E * H_HEADS);
        k_edge_exp<<<cdiv(E * H_HEADS, TB), TB, 0, stream>>>(dst, mmax, eArr, den, E * H_HEADS);
        k_edge_aggregate<<<cdiv(E * 32, TB), TB, 0, stream>>>(src, dst, eArr, feat, rst, E);
        k_post_norm<<<cdiv(N, 8), TB, 0, stream>>>(rst, den, hres, gl[l], bl[l], hres, hbf, pool, l, N);
    }

    k_final<<<cdiv(Bg * 192, TB), TB, 0, stream>>>(pool, out, Bg * 192);
}
